// SIR_mamba_42545946034890
// MI455X (gfx1250) — compile-verified, matrix operand order checked by reading, not run on hardware
//
#include <hip/hip_runtime.h>

// ---------------------------------------------------------------------------
// SIR-GCN block for MI455X (gfx1250, wave32, WMMA + async-to-LDS staging).
// GEMMs: v_wmma_f32_16x16x32_bf16 with fp32 accumulate. Weights re-packed
// once per call into the WMMA B-fragment lane layout (32 contiguous B/lane).
// Activation tiles are staged global->LDS with GLOBAL_LOAD_ASYNC_TO_LDS_B128
// (ASYNCcnt path), fp32 in LDS, converted to bf16 at A-fragment load time.
// Scatter-add uses global fp32 atomics. Workspace: ~83 MB.
// ---------------------------------------------------------------------------

#define BATCH 4
#define NNODE 10000
#define DIM   128
#define NEDGE 160000
#define ROWS  (BATCH * NNODE)     // 40000
#define TILE_M 64
#define NTILES (ROWS / TILE_M)    // 625 exactly
#define SLDA   132                // fp32 stage tile leading dim (pad vs 64 banks)

typedef __attribute__((ext_vector_type(16))) __bf16 v16bf;
typedef __attribute__((ext_vector_type(8)))  __bf16 bf16x8;
typedef __attribute__((ext_vector_type(8)))  float  v8f;

union FragB { v16bf v; bf16x8 h[2]; };

// packed-weight offsets (bf16 elements) inside workspace
#define PWS   0
#define PWI   16384
#define PWR   32768
#define PTOI  49152
#define PTOR  81920
#define POUT  98304
#define PFF1  147456
#define PFF2  180224

#define WMMA_BF16(A, B, C) \
  __builtin_amdgcn_wmma_f32_16x16x32_bf16(false, (A), false, (B), (short)0, (C), false, false)

// ---- async stage: 64x128 fp32 tile, global -> LDS (blockDim == 256) -------
// Each lane issues 8 x 16B async copies; LDS written directly (no VGPR data).
__device__ __forceinline__ void stage_tile_async(const float* __restrict__ g,
                                                 int rowBase, float* lds) {
  #pragma unroll
  for (int t = 0; t < 8; ++t) {
    const int idx = (int)threadIdx.x + t * 256;       // 0..2047
    const int r = idx >> 5, c4 = idx & 31;            // 32 float4 per row
    const float* src = g + (size_t)(rowBase + r) * DIM + c4 * 4;
    const unsigned dst = (unsigned)(uintptr_t)(lds + r * SLDA + c4 * 4); // LDS byte addr
    asm volatile("global_load_async_to_lds_b128 %0, %1, off"
                 :: "v"(dst), "v"((unsigned long long)(uintptr_t)src)
                 : "memory");
  }
}

__device__ __forceinline__ void wait_async_and_barrier() {
  asm volatile("s_wait_asynccnt 0x0" ::: "memory");
  __syncthreads();
}

// -------- A fragment: 16x32 tile from fp32 LDS, converted to bf16 ----------
// lanes 0-15: M=lane, elems 0..7 = K..K+7, elems 8..15 = K+16..K+23
// lanes16-31: M=lane-16, elems 0..7 = K+8..K+15, elems 8..15 = K+24..K+31
__device__ __forceinline__ v16bf load_a_f32(const float* lds, int lda, int mBase,
                                            int kBase, int lane) {
  const int m  = mBase + (lane & 15);
  const int kb = kBase + ((lane >> 4) << 3);
  const float* p = lds + m * lda + kb;
  v16bf a;
  #pragma unroll
  for (int e = 0; e < 8; ++e) a[e] = (__bf16)p[e];
  #pragma unroll
  for (int e = 0; e < 8; ++e) a[8 + e] = (__bf16)p[16 + e];
  return a;
}

// -------- A fragment from bf16 LDS (concat / hidden buffers) ---------------
__device__ __forceinline__ v16bf load_a_bf16(const __bf16* lds, int lda, int mBase,
                                             int kBase, int lane) {
  const int m  = mBase + (lane & 15);
  const int kb = kBase + ((lane >> 4) << 3);
  const __bf16* p = lds + m * lda + kb;
  FragB f;
  f.h[0] = *(const bf16x8*)(p);
  f.h[1] = *(const bf16x8*)(p + 16);
  return f.v;
}

// -------- B fragment: pre-packed, 32 bytes contiguous per lane -------------
__device__ __forceinline__ v16bf load_b(const __bf16* __restrict__ P, int ntn,
                                        int kt, int nt, int lane) {
  const __bf16* p = P + ((size_t)((kt * ntn + nt) * 32 + lane)) * 16;
  FragB f;
  f.h[0] = *(const bf16x8*)(p);
  f.h[1] = *(const bf16x8*)(p + 8);
  return f.v;
}

// ===========================================================================
// K0: pack one fp32 weight [Kdim x Ndim] row-major -> WMMA B-fragment layout
// ===========================================================================
__global__ void pack_w_kernel(const float* __restrict__ W, __bf16* __restrict__ P,
                              int Kdim, int Ndim) {
  const int t = blockIdx.x * blockDim.x + threadIdx.x;
  const int ntn = Ndim >> 4, ntk = Kdim >> 5;
  const int total = ntk * ntn * 32;
  if (t >= total) return;
  const int lane = t & 31;
  const int tile = t >> 5;
  const int nt = tile % ntn, kt = tile / ntn;
  const int n = (nt << 4) + (lane & 15);
  const int kbase = (kt << 5) + ((lane >> 4) << 3);
  __bf16* dst = P + ((size_t)tile * 32 + lane) * 16;
  #pragma unroll
  for (int grp = 0; grp < 2; ++grp) {
    #pragma unroll
    for (int e = 0; e < 8; ++e) {
      const int k = kbase + grp * 16 + e;
      dst[grp * 8 + e] = (__bf16)W[(size_t)k * Ndim + n];
    }
  }
}

// ===========================================================================
// K1: s = relu(x@Ws+bs+x), i = relu(x@Wi+bi+x), r = relu(x@Wr+br+x)
// ===========================================================================
__global__ __launch_bounds__(256) void proj_kernel(
    const float* __restrict__ x, const __bf16* __restrict__ pw,
    const float* __restrict__ bs, const float* __restrict__ bi,
    const float* __restrict__ br,
    float* __restrict__ so, float* __restrict__ io, float* __restrict__ ro) {
  __shared__ __align__(16) float xs[TILE_M * SLDA];   // 33792 B
  const int rowBase = blockIdx.x * TILE_M;
  stage_tile_async(x, rowBase, xs);
  wait_async_and_barrier();

  const int wave = threadIdx.x >> 5, lane = threadIdx.x & 31;
  const int mB = (wave & 3) << 4;        // 0,16,32,48
  const int nH = (wave >> 2) << 6;       // 0 or 64

  v16bf af[4];
  #pragma unroll
  for (int kt = 0; kt < 4; ++kt) af[kt] = load_a_f32(xs, SLDA, mB, kt << 5, lane);

  const float* bias[3] = {bs, bi, br};
  float* outp[3] = {so, io, ro};
  const int rOff = (lane >> 4) << 3;

  #pragma unroll
  for (int m = 0; m < 3; ++m) {
    const __bf16* P = pw + m * 16384;
    v8f acc[4] = {};
    #pragma unroll
    for (int nt = 0; nt < 4; ++nt)
      #pragma unroll
      for (int kt = 0; kt < 4; ++kt)
        acc[nt] = WMMA_BF16(af[kt], load_b(P, 8, kt, (nH >> 4) + nt, lane), acc[nt]);
    #pragma unroll
    for (int nt = 0; nt < 4; ++nt) {
      const int col = nH + (nt << 4) + (lane & 15);
      const float bc = bias[m][col];
      #pragma unroll
      for (int v = 0; v < 8; ++v) {
        const int g = rowBase + mB + rOff + v;
        const float val = acc[nt][v] + bc + x[(size_t)g * DIM + col];
        outp[m][(size_t)g * DIM + col] = fmaxf(val, 0.0f);
      }
    }
  }
}

// ===========================================================================
// K2: neighbor_i[:, dst, :] += i[:, src, :] * w   (fp32 atomics)
// ===========================================================================
__global__ __launch_bounds__(256) void scatter_kernel(
    const float* __restrict__ iM, const int* __restrict__ ei,
    const float* __restrict__ w, float* __restrict__ nb) {
  const int t = blockIdx.x * blockDim.x + threadIdx.x;  // E * 32 threads
  if (t >= NEDGE * 32) return;
  const int e  = t >> 5;
  const int c4 = t & 31;
  const int sN = ei[e], dN = ei[NEDGE + e];
  const float we = w[e];
  #pragma unroll
  for (int b = 0; b < BATCH; ++b) {
    const float4 v = *(const float4*)(iM + ((size_t)b * NNODE + sN) * DIM + c4 * 4);
    float* o = nb + ((size_t)b * NNODE + dN) * DIM + c4 * 4;
    atomicAdd(o + 0, v.x * we);
    atomicAdd(o + 1, v.y * we);
    atomicAdd(o + 2, v.z * we);
    atomicAdd(o + 3, v.w * we);
  }
}

// ===========================================================================
// K3: toI, toR, s1/i1/r1, x_SIR = cat(s1,i1,r1)@Wout + bout + x -> d_out
// Shared buffer time-aliased: fp32 stage tile (64x132) <-> bf16 concat (64x392)
// ===========================================================================
__global__ __launch_bounds__(256) void mid_kernel(
    const float* __restrict__ x, const float* __restrict__ sbuf,
    const float* __restrict__ ibuf, const float* __restrict__ rbuf,
    const float* __restrict__ nbuf, const __bf16* __restrict__ pw,
    const float* __restrict__ btoI, const float* __restrict__ btoR,
    const float* __restrict__ bout, float* __restrict__ out) {
  __shared__ __align__(16) char smraw[TILE_M * 392 * 2];  // 50176 B
  float*  stg = (float*)smraw;        // phase 1: fp32 stage tile, lda SLDA
  __bf16* cat = (__bf16*)smraw;       // phase 2: 64x384 concat, lda 392

  const int rowBase = blockIdx.x * TILE_M;
  const int wave = threadIdx.x >> 5, lane = threadIdx.x & 31;
  const int mB = (wave & 3) << 4, nH = (wave >> 2) << 6;
  const int rOff = (lane >> 4) << 3;

  v8f accI[4] = {}, accR[4] = {};

  // toI += s @ WtoI[0:128, :]
  stage_tile_async(sbuf, rowBase, stg);
  wait_async_and_barrier();
  #pragma unroll
  for (int kt = 0; kt < 4; ++kt) {
    const v16bf af = load_a_f32(stg, SLDA, mB, kt << 5, lane);
    #pragma unroll
    for (int nt = 0; nt < 4; ++nt)
      accI[nt] = WMMA_BF16(af, load_b(pw + PTOI, 8, kt, (nH >> 4) + nt, lane), accI[nt]);
  }
  __syncthreads();

  // toI += neighbor @ WtoI[128:256, :]
  stage_tile_async(nbuf, rowBase, stg);
  wait_async_and_barrier();
  #pragma unroll
  for (int kt = 0; kt < 4; ++kt) {
    const v16bf af = load_a_f32(stg, SLDA, mB, kt << 5, lane);
    #pragma unroll
    for (int nt = 0; nt < 4; ++nt)
      accI[nt] = WMMA_BF16(af, load_b(pw + PTOI, 8, kt + 4, (nH >> 4) + nt, lane), accI[nt]);
  }
  __syncthreads();

  // toR = i @ WtoR
  stage_tile_async(ibuf, rowBase, stg);
  wait_async_and_barrier();
  #pragma unroll
  for (int kt = 0; kt < 4; ++kt) {
    const v16bf af = load_a_f32(stg, SLDA, mB, kt << 5, lane);
    #pragma unroll
    for (int nt = 0; nt < 4; ++nt)
      accR[nt] = WMMA_BF16(af, load_b(pw + PTOR, 8, kt, (nH >> 4) + nt, lane), accR[nt]);
  }
  __syncthreads();  // done reading stage region; reuse as concat buffer

  // build s1 | i1 | r1 into LDS (bf16) for the Wout GEMM
  #pragma unroll
  for (int nt = 0; nt < 4; ++nt) {
    const int col = nH + (nt << 4) + (lane & 15);
    const float bI = btoI[col], bR = btoR[col];
    #pragma unroll
    for (int v = 0; v < 8; ++v) {
      const int lr = mB + rOff + v;
      const int g  = rowBase + lr;
      const float toI = accI[nt][v] + bI;
      const float toR = accR[nt][v] + bR;
      const float sv = sbuf[(size_t)g * DIM + col];
      const float iv = ibuf[(size_t)g * DIM + col];
      const float rv = rbuf[(size_t)g * DIM + col];
      cat[lr * 392 + col]       = (__bf16)(sv - toI);
      cat[lr * 392 + 128 + col] = (__bf16)(iv + toI - toR);
      cat[lr * 392 + 256 + col] = (__bf16)(toR + rv);
    }
  }
  __syncthreads();

  // x_SIR = cat @ Wout + bout + x   (K = 384)
  v8f acc[4] = {};
  #pragma unroll
  for (int kt = 0; kt < 12; ++kt) {
    const v16bf af = load_a_bf16(cat, 392, mB, kt << 5, lane);
    #pragma unroll
    for (int nt = 0; nt < 4; ++nt)
      acc[nt] = WMMA_BF16(af, load_b(pw + POUT, 8, kt, (nH >> 4) + nt, lane), acc[nt]);
  }
  #pragma unroll
  for (int nt = 0; nt < 4; ++nt) {
    const int col = nH + (nt << 4) + (lane & 15);
    const float bo = bout[col];
    #pragma unroll
    for (int v = 0; v < 8; ++v) {
      const int g = rowBase + mB + rOff + v;
      out[(size_t)g * DIM + col] = acc[nt][v] + bo + x[(size_t)g * DIM + col];
    }
  }
}

// ===========================================================================
// K4: out += relu(out@Wff1 + bff1) @ Wff2 + bff2
// Shared buffer time-aliased: fp32 x_SIR stage (64x132) <-> bf16 hidden (64x264)
// ===========================================================================
__global__ __launch_bounds__(256) void ff_kernel(
    const __bf16* __restrict__ pw, const float* __restrict__ bff1,
    const float* __restrict__ bff2, float* __restrict__ out) {
  __shared__ __align__(16) char smraw[TILE_M * 264 * 2];  // 33792 B
  float*  stg = (float*)smraw;       // phase 1: fp32 x_SIR tile, lda SLDA
  __bf16* hid = (__bf16*)smraw;      // phase 2: 64x256 hidden, lda 264

  const int rowBase = blockIdx.x * TILE_M;
  stage_tile_async(out, rowBase, stg);
  wait_async_and_barrier();

  const int wave = threadIdx.x >> 5, lane = threadIdx.x & 31;
  const int mB = (wave & 3) << 4;
  const int rOff = (lane >> 4) << 3;

  v16bf af[4];
  #pragma unroll
  for (int kt = 0; kt < 4; ++kt) af[kt] = load_a_f32(stg, SLDA, mB, kt << 5, lane);

  // hidden = relu(x_SIR @ Wff1 + bff1), N = 256
  const int nH1 = (wave >> 2) << 7;  // 0 or 128
  v8f acc[8] = {};
  #pragma unroll
  for (int nt = 0; nt < 8; ++nt)
    #pragma unroll
    for (int kt = 0; kt < 4; ++kt)
      acc[nt] = WMMA_BF16(af[kt], load_b(pw + PFF1, 16, kt, (nH1 >> 4) + nt, lane), acc[nt]);
  __syncthreads();  // done reading stage region
  #pragma unroll
  for (int nt = 0; nt < 8; ++nt) {
    const int col = nH1 + (nt << 4) + (lane & 15);
    const float b1 = bff1[col];
    #pragma unroll
    for (int v = 0; v < 8; ++v) {
      const int lr = mB + rOff + v;
      hid[lr * 264 + col] = (__bf16)fmaxf(acc[nt][v] + b1, 0.0f);
    }
  }
  __syncthreads();

  // out = hidden @ Wff2 + bff2 + x_SIR   (K = 256)
  const int nH2 = (wave >> 2) << 6;
  v8f acc2[4] = {};
  #pragma unroll
  for (int kt = 0; kt < 8; ++kt) {
    const v16bf a2 = load_a_bf16(hid, 264, mB, kt << 5, lane);
    #pragma unroll
    for (int nt = 0; nt < 4; ++nt)
      acc2[nt] = WMMA_BF16(a2, load_b(pw + PFF2, 8, kt, (nH2 >> 4) + nt, lane), acc2[nt]);
  }
  #pragma unroll
  for (int nt = 0; nt < 4; ++nt) {
    const int col = nH2 + (nt << 4) + (lane & 15);
    const float b2 = bff2[col];
    #pragma unroll
    for (int v = 0; v < 8; ++v) {
      const int g = rowBase + mB + rOff + v;
      out[(size_t)g * DIM + col] = acc2[nt][v] + b2 + out[(size_t)g * DIM + col];
    }
  }
}

// ===========================================================================
extern "C" void kernel_launch(void* const* d_in, const int* in_sizes, int n_in,
                              void* d_out, int out_size, void* d_ws, size_t ws_size,
                              hipStream_t stream) {
  (void)in_sizes; (void)n_in; (void)out_size; (void)ws_size;

  const float* x    = (const float*)d_in[0];
  const int*   ei   = (const int*)d_in[1];
  const float* w    = (const float*)d_in[2];
  const float* Ws   = (const float*)d_in[3];  const float* bs   = (const float*)d_in[4];
  const float* Wi   = (const float*)d_in[5];  const float* bi   = (const float*)d_in[6];
  const float* Wr   = (const float*)d_in[7];  const float* br   = (const float*)d_in[8];
  const float* WtoI = (const float*)d_in[9];  const float* btoI = (const float*)d_in[10];
  const float* WtoR = (const float*)d_in[11]; const float* btoR = (const float*)d_in[12];
  const float* Wout = (const float*)d_in[13]; const float* bout = (const float*)d_in[14];
  const float* Wff1 = (const float*)d_in[15]; const float* bff1 = (const float*)d_in[16];
  const float* Wff2 = (const float*)d_in[17]; const float* bff2 = (const float*)d_in[18];
  float* out = (float*)d_out;

  // workspace layout
  __bf16* pw  = (__bf16*)d_ws;                              // 212992 bf16 packed weights
  char* base  = (char*)d_ws + (1 << 19);                    // 512 KB aligned region
  const size_t TSZ = (size_t)ROWS * DIM;                    // 5,120,000 floats
  float* sbuf = (float*)base;
  float* ibuf = sbuf + TSZ;
  float* rbuf = ibuf + TSZ;
  float* nbuf = rbuf + TSZ;                                 // total ws use ~83 MB

  // K0: pack all weights to WMMA B-fragment bf16 layout
  auto pack = [&](const float* W, int off, int Kd, int Nd) {
    const int total = (Kd >> 5) * (Nd >> 4) * 32;
    pack_w_kernel<<<(total + 255) / 256, 256, 0, stream>>>(W, pw + off, Kd, Nd);
  };
  pack(Ws,   PWS,  128, 128);
  pack(Wi,   PWI,  128, 128);
  pack(Wr,   PWR,  128, 128);
  pack(WtoI, PTOI, 256, 128);
  pack(WtoR, PTOR, 128, 128);
  pack(Wout, POUT, 384, 128);
  pack(Wff1, PFF1, 128, 256);
  pack(Wff2, PFF2, 256, 128);

  // K1: input projections s, i, r
  proj_kernel<<<NTILES, 256, 0, stream>>>(x, pw, bs, bi, br, sbuf, ibuf, rbuf);

  // K2: zero + scatter-add neighbor_i
  hipMemsetAsync(nbuf, 0, TSZ * sizeof(float), stream);
  scatter_kernel<<<(NEDGE * 32) / 256, 256, 0, stream>>>(ibuf, ei, w, nbuf);

  // K3: toI/toR + SIR update + Wout GEMM -> x_SIR in d_out
  mid_kernel<<<NTILES, 256, 0, stream>>>(x, sbuf, ibuf, rbuf, nbuf, pw,
                                         btoI, btoR, bout, out);

  // K4: feed-forward block, fused in LDS -> final output in d_out
  ff_kernel<<<NTILES, 256, 0, stream>>>(pw, bff1, bff2, out);
}
